// SimpleNTLBGModel_52132313039175
// MI455X (gfx1250) — matrix operand order repaired and checked by reading, not compiled, not run on hardware
//
#include <hip/hip_runtime.h>
#include <hip/hip_bf16.h>

// ---------------------------------------------------------------------------
// SimpleNTLBGModel forward on MI455X (gfx1250, wave32, WMMA bf16)
//
// Pipeline:
//   1) enc_score : scores[b,t] = relu(vf@W_enc+b_enc) . W_sel + b_sel  (fused,
//                  ve never materialized; WMMA bf16, fp32 accumulate)
//   2) topk      : top-6 frame indices per batch
//   3) reps      : recompute encoder rows for the 6 winners -> reps (+d_out)
//   4) kv        : k = reps@W_k+b_k, v = reps@W_v+b_v
//   5) q         : q = emb[ids]@W_q+b_q        (gathered-A WMMA GEMM)
//   6) woc/boc   : W_oc = W_o@W_cls (padded to 1024 cols), b_oc fold
//   7) attn      : softmax over K=6 keys, ctx (tiny, VALU)
//   8) logits    : ctx @ W_oc + b_oc -> d_out  (WMMA, branch-free via pad)
// ---------------------------------------------------------------------------

typedef __attribute__((ext_vector_type(16))) __bf16 v16bf;
typedef __attribute__((ext_vector_type(8)))  float  v8f;
typedef __attribute__((ext_vector_type(4)))  float  v4f;

#define BB    32
#define TT    4096
#define LL    512
#define DIN   768
#define DD    512
#define KK    6
#define NH    8
#define HDIM  64
#define VOC   1000
#define WOCLD 1024   // padded leading dim for W_oc in workspace

// fp32 -> bf16; plain conversion so the backend can emit packed v_cvt (RNE)
__device__ inline __bf16 f2bf(float f) { return (__bf16)f; }

// convert 8 consecutive fp32 -> dst[base..base+7]
__device__ inline void cvt8(v16bf& dst, int base, const float* __restrict__ p) {
  v4f x0 = *(const v4f*)p;
  v4f x1 = *(const v4f*)(p + 4);
  dst[base + 0] = f2bf(x0.x); dst[base + 1] = f2bf(x0.y);
  dst[base + 2] = f2bf(x0.z); dst[base + 3] = f2bf(x0.w);
  dst[base + 4] = f2bf(x1.x); dst[base + 5] = f2bf(x1.y);
  dst[base + 6] = f2bf(x1.z); dst[base + 7] = f2bf(x1.w);
}

// B tile column for this lane: 16 K values (table order), stride ldb. NEDGE=false -> branch-free.
template <bool NEDGE>
__device__ inline v16bf loadB(const float* __restrict__ Bk, int ldb, int col, int nlim) {
  v16bf b;
  const float* p = Bk + col;
  const bool ok = !NEDGE || col < nlim;
#pragma unroll
  for (int e = 0; e < 16; ++e) {
    int k = (e < 8) ? e : e + 8;
    b[e] = ok ? f2bf(p[(size_t)k * ldb]) : f2bf(0.0f);
  }
  return b;
}

template <int MT, int NT>
__device__ inline void zero_acc(v8f (&acc)[MT][NT]) {
#pragma unroll
  for (int mt = 0; mt < MT; ++mt)
#pragma unroll
    for (int nt = 0; nt < NT; ++nt)
#pragma unroll
      for (int r = 0; r < 8; ++r) acc[mt][nt][r] = 0.0f;
}

// Core K-loop: acc += bf16(A) x bf16(B), fp32 accumulate.
// A layout (16x32, MxK): lane m=l&15; half h=l>>4:
//   elems 0..7  <- K = kb + 8h + 0..7
//   elems 8..15 <- K = kb + 16 + 8h + 0..7
// B layout (32x16, KxN): lane supplies col N = l&15, elems = K in same order as A.
template <int MT, int NT, bool NEDGE>
__device__ inline void wmma_kloop(const float* const* arow,            // per-lane A row pointers
                                  const float* __restrict__ B, int ldb,
                                  int nbase, int nlim, int Ksz,
                                  v8f (&acc)[MT][NT]) {
  const int lane = threadIdx.x & 31;
  const int half = lane >> 4;
  for (int kb = 0; kb < Ksz; kb += 32) {
    v16bf a[MT];
#pragma unroll
    for (int mt = 0; mt < MT; ++mt) {
      const float* p = arow[mt] + kb + half * 8;
      cvt8(a[mt], 0, p);
      cvt8(a[mt], 8, p + 16);
      // speculative prefetch of the next K-step; silently dropped past end
      __builtin_prefetch(p + 32, 0, 3);
    }
    const float* Brow = B + (size_t)(kb + half * 8) * ldb;
#pragma unroll
    for (int nt = 0; nt < NT; ++nt) {
      v16bf b = loadB<NEDGE>(Brow, ldb, nbase + nt * 16 + (lane & 15), nlim);
#pragma unroll
      for (int mt = 0; mt < MT; ++mt)
        acc[mt][nt] = __builtin_amdgcn_wmma_f32_16x16x32_bf16(
            false, a[mt], false, b, (short)0, acc[mt][nt], false, false);
    }
  }
}

// C/D 16x16 f32: VGPR r, lanes 0-15 -> M=r, lanes 16-31 -> M=8+r; N = lane&15.
// zero_pad: store 0.0f for cols in [nlim, ld-pad) instead of skipping (used to
// zero-fill the padded W_oc columns so the logits pass can be branch-free).
template <int MT, int NT>
__device__ inline void store_tile(float* __restrict__ C, int ldc,
                                  int mbase, int mlim, int nbase, int nlim,
                                  const float* __restrict__ bias, bool relu,
                                  bool zero_pad, v8f (&acc)[MT][NT]) {
  const int lane  = threadIdx.x & 31;
  const int half  = lane >> 4;
  const int nlane = lane & 15;
#pragma unroll
  for (int mt = 0; mt < MT; ++mt)
#pragma unroll
    for (int r = 0; r < 8; ++r) {
      int row = mbase + mt * 16 + half * 8 + r;
      if (row >= mlim) continue;
#pragma unroll
      for (int nt = 0; nt < NT; ++nt) {
        int col = nbase + nt * 16 + nlane;
        if (col >= nlim) {
          if (zero_pad) C[(size_t)row * ldc + col] = 0.0f;
          continue;
        }
        float v = acc[mt][nt][r];
        if (bias) v += bias[col];
        if (relu) v = fmaxf(v, 0.0f);
        C[(size_t)row * ldc + col] = v;
      }
    }
}

// --------------------------------------------------------------------------
// 1) Fused encoder GEMM + frame scores. Block: 256 thr (8 waves), 64 rows.
//    Wave w owns N-range [64w, 64w+64). ve tile stays in WMMA accumulators.
// --------------------------------------------------------------------------
__global__ __launch_bounds__(256) void enc_score_kernel(
    const float* __restrict__ vf, const float* __restrict__ W_enc,
    const float* __restrict__ b_enc, const float* __restrict__ W_sel,
    const float* __restrict__ b_sel, float* __restrict__ scores) {
  __shared__ float part[8][64];
  const int wave  = threadIdx.x >> 5;
  const int lane  = threadIdx.x & 31;
  const int half  = lane >> 4;
  const int nlane = lane & 15;
  const int rowbase = blockIdx.x * 64;

  const float* arow[4];
#pragma unroll
  for (int mt = 0; mt < 4; ++mt)
    arow[mt] = vf + (size_t)(rowbase + mt * 16 + nlane) * DIN;

  v8f acc[4][4];
  zero_acc<4, 4>(acc);
  const int nbase = wave * 64;
  wmma_kloop<4, 4, false>(arow, W_enc, DD, nbase, DD, DIN, acc);

  // score partials: p[mt][r] = sum_n relu(ve+b) * W_sel[n] over this wave's N
  float p[4][8];
#pragma unroll
  for (int mt = 0; mt < 4; ++mt)
#pragma unroll
    for (int r = 0; r < 8; ++r) p[mt][r] = 0.0f;

#pragma unroll
  for (int nt = 0; nt < 4; ++nt) {
    int col  = nbase + nt * 16 + nlane;
    float be = b_enc[col];
    float ws = W_sel[col];
#pragma unroll
    for (int mt = 0; mt < 4; ++mt)
#pragma unroll
      for (int r = 0; r < 8; ++r) {
        float v = fmaxf(acc[mt][nt][r] + be, 0.0f);
        p[mt][r] += v * ws;
      }
  }
  // reduce across the 16 lanes of each half (xor 1,2,4,8 stays inside half)
#pragma unroll
  for (int mt = 0; mt < 4; ++mt)
#pragma unroll
    for (int r = 0; r < 8; ++r) {
      float v = p[mt][r];
      v += __shfl_xor(v, 1, 32);
      v += __shfl_xor(v, 2, 32);
      v += __shfl_xor(v, 4, 32);
      v += __shfl_xor(v, 8, 32);
      p[mt][r] = v;
    }
  if (nlane == 0) {
#pragma unroll
    for (int mt = 0; mt < 4; ++mt)
#pragma unroll
      for (int r = 0; r < 8; ++r)
        part[wave][mt * 16 + half * 8 + r] = p[mt][r];
  }
  __syncthreads();
  if (threadIdx.x < 64) {
    float s = b_sel[0];
#pragma unroll
    for (int w = 0; w < 8; ++w) s += part[w][threadIdx.x];
    scores[rowbase + threadIdx.x] = s;
  }
}

// --------------------------------------------------------------------------
// 2) top-6 per batch (value desc, lowest index on ties, matching lax.top_k)
// --------------------------------------------------------------------------
__global__ __launch_bounds__(256) void topk_kernel(
    const float* __restrict__ scores, int* __restrict__ top_idx) {
  __shared__ float sv[TT];
  __shared__ float rv[256];
  __shared__ int   ri[256];
  const int b = blockIdx.x;
  for (int i = threadIdx.x; i < TT; i += 256) sv[i] = scores[b * TT + i];
  __syncthreads();
  for (int kk = 0; kk < KK; ++kk) {
    float best = -3.0e38f;
    int   bi   = 0x7FFFFFFF;
    for (int i = threadIdx.x; i < TT; i += 256) {
      float v = sv[i];
      if (v > best || (v == best && i < bi)) { best = v; bi = i; }
    }
    rv[threadIdx.x] = best;
    ri[threadIdx.x] = bi;
    __syncthreads();
    for (int s = 128; s > 0; s >>= 1) {
      if (threadIdx.x < s) {
        float ov = rv[threadIdx.x + s];
        int   oi = ri[threadIdx.x + s];
        if (ov > rv[threadIdx.x] || (ov == rv[threadIdx.x] && oi < ri[threadIdx.x])) {
          rv[threadIdx.x] = ov;
          ri[threadIdx.x] = oi;
        }
      }
      __syncthreads();
    }
    if (threadIdx.x == 0) {
      top_idx[b * KK + kk] = ri[0];
      sv[ri[0]] = -3.0e38f;   // mask winner for next round
    }
    __syncthreads();
  }
}

// --------------------------------------------------------------------------
// 3) reps: recompute encoder rows for the 192 selected frames (WMMA).
//    grid (12, 8), one wave per block: 16 rows x 64 cols.
// --------------------------------------------------------------------------
__global__ __launch_bounds__(32) void reps_kernel(
    const float* __restrict__ vf, const float* __restrict__ W_enc,
    const float* __restrict__ b_enc, const int* __restrict__ top_idx,
    float* __restrict__ reps_ws, float* __restrict__ reps_out) {
  const int lane  = threadIdx.x & 31;
  const int half  = lane >> 4;
  const int nlane = lane & 15;
  const int rg = blockIdx.x * 16 + nlane;        // 0..191
  const int b  = rg / KK;
  const int kk = rg % KK;
  const int t  = top_idx[b * KK + kk];
  const float* arow[1] = { vf + ((size_t)b * TT + t) * DIN };

  v8f acc[1][4];
  zero_acc<1, 4>(acc);
  const int nbase = blockIdx.y * 64;
  wmma_kloop<1, 4, false>(arow, W_enc, DD, nbase, DD, DIN, acc);

#pragma unroll
  for (int r = 0; r < 8; ++r) {
    int row = blockIdx.x * 16 + half * 8 + r;
#pragma unroll
    for (int nt = 0; nt < 4; ++nt) {
      int col = nbase + nt * 16 + nlane;
      float v = fmaxf(acc[0][nt][r] + b_enc[col], 0.0f);
      reps_ws[(size_t)row * DD + col]  = v;
      reps_out[(size_t)row * DD + col] = v;
    }
  }
}

// --------------------------------------------------------------------------
// 4) k / v projections of reps. grid (12, 8, 2), one wave per block.
// --------------------------------------------------------------------------
__global__ __launch_bounds__(32) void kv_kernel(
    const float* __restrict__ reps, const float* __restrict__ W_k,
    const float* __restrict__ b_k, const float* __restrict__ W_v,
    const float* __restrict__ b_v, float* __restrict__ kmat,
    float* __restrict__ vmat) {
  const int lane  = threadIdx.x & 31;
  const int nlane = lane & 15;
  const int z = blockIdx.z;
  const float* W    = z ? W_v : W_k;
  const float* bias = z ? b_v : b_k;
  float*       outp = z ? vmat : kmat;

  const float* arow[1] = { reps + (size_t)(blockIdx.x * 16 + nlane) * DD };
  v8f acc[1][4];
  zero_acc<1, 4>(acc);
  const int nbase = blockIdx.y * 64;
  wmma_kloop<1, 4, false>(arow, W, DD, nbase, DD, DD, acc);
  store_tile<1, 4>(outp, DD, blockIdx.x * 16, BB * KK, nbase, DD, bias, false,
                   false, acc);
}

// --------------------------------------------------------------------------
// 5) q = emb[input_ids] @ W_q + b_q. grid 256, block 256 (8 waves, 64 rows).
// --------------------------------------------------------------------------
__global__ __launch_bounds__(256) void q_kernel(
    const int* __restrict__ input_ids, const float* __restrict__ emb,
    const float* __restrict__ W_q, const float* __restrict__ b_q,
    float* __restrict__ qbuf) {
  const int wave  = threadIdx.x >> 5;
  const int lane  = threadIdx.x & 31;
  const int nlane = lane & 15;
  const int rowbase = blockIdx.x * 64;

  const float* arow[4];
#pragma unroll
  for (int mt = 0; mt < 4; ++mt) {
    int row = rowbase + mt * 16 + nlane;
    arow[mt] = emb + (size_t)input_ids[row] * DD;   // gathered A
  }
  v8f acc[4][4];
  zero_acc<4, 4>(acc);
  const int nbase = wave * 64;
  wmma_kloop<4, 4, false>(arow, W_q, DD, nbase, DD, DD, acc);
  store_tile<4, 4>(qbuf, DD, rowbase, BB * LL, nbase, DD, b_q, false, false, acc);
}

// --------------------------------------------------------------------------
// 6) Fold output projection into classifier: W_oc = W_o @ W_cls.
//    Stored padded [512][1024]; cols 1000..1023 zero-filled so the big logits
//    pass can run branch-free. grid (8, 2), block 256.
// --------------------------------------------------------------------------
__global__ __launch_bounds__(256) void woc_kernel(
    const float* __restrict__ W_o, const float* __restrict__ W_cls,
    float* __restrict__ Woc) {
  const int wave  = threadIdx.x >> 5;
  const int lane  = threadIdx.x & 31;
  const int nlane = lane & 15;
  const int rowbase = blockIdx.x * 64;

  const float* arow[4];
#pragma unroll
  for (int mt = 0; mt < 4; ++mt)
    arow[mt] = W_o + (size_t)(rowbase + mt * 16 + nlane) * DD;

  v8f acc[4][4];
  zero_acc<4, 4>(acc);
  const int nbase = blockIdx.y * 512 + wave * 64;
  wmma_kloop<4, 4, true>(arow, W_cls, VOC, nbase, VOC, DD, acc);
  store_tile<4, 4>(Woc, WOCLD, rowbase, DD, nbase, VOC, nullptr, false,
                   /*zero_pad=*/true, acc);
}

__global__ __launch_bounds__(256) void boc_kernel(
    const float* __restrict__ b_o, const float* __restrict__ W_cls,
    const float* __restrict__ b_cls, float* __restrict__ boc) {
  int n = blockIdx.x * 256 + threadIdx.x;
  if (n < VOC) {
    float s = b_cls[n];
    for (int d = 0; d < DD; ++d) s += b_o[d] * W_cls[(size_t)d * VOC + n];
    boc[n] = s;
  }
}

// --------------------------------------------------------------------------
// 7) attention over K=6 keys. grid (B*L, NH), one wave; lane owns 2 dims.
// --------------------------------------------------------------------------
__global__ __launch_bounds__(32) void attn_kernel(
    const float* __restrict__ qbuf, const float* __restrict__ kmat,
    const float* __restrict__ vmat, float* __restrict__ ctx) {
  const int lane = threadIdx.x & 31;
  const int row  = blockIdx.x;           // b*L + l
  const int b    = row / LL;
  const int h    = blockIdx.y;
  const int d0   = lane * 2;

  const float* qp = qbuf + (size_t)row * DD + h * HDIM;
  float q0 = qp[d0], q1 = qp[d0 + 1];

  float s[KK];
#pragma unroll
  for (int kk = 0; kk < KK; ++kk) {
    const float* kp = kmat + (size_t)(b * KK + kk) * DD + h * HDIM;
    float p = q0 * kp[d0] + q1 * kp[d0 + 1];
    p += __shfl_xor(p, 1, 32);
    p += __shfl_xor(p, 2, 32);
    p += __shfl_xor(p, 4, 32);
    p += __shfl_xor(p, 8, 32);
    p += __shfl_xor(p, 16, 32);
    s[kk] = p * 0.125f;                  // 1/sqrt(64)
  }
  float m = s[0];
#pragma unroll
  for (int kk = 1; kk < KK; ++kk) m = fmaxf(m, s[kk]);
  float den = 0.0f;
#pragma unroll
  for (int kk = 0; kk < KK; ++kk) { s[kk] = __expf(s[kk] - m); den += s[kk]; }
  float inv = 1.0f / den;

  float c0 = 0.0f, c1 = 0.0f;
#pragma unroll
  for (int kk = 0; kk < KK; ++kk) {
    const float* vp = vmat + (size_t)(b * KK + kk) * DD + h * HDIM;
    float a = s[kk] * inv;
    c0 += a * vp[d0];
    c1 += a * vp[d0 + 1];
  }
  float* cp = ctx + (size_t)row * DD + h * HDIM;
  cp[d0]     = c0;
  cp[d0 + 1] = c1;
}

// --------------------------------------------------------------------------
// 8) logits = ctx @ W_oc + b_oc. grid (256, 2), block 256.
//    B is the padded [512][1024] W_oc -> branch-free K-loop; store guards
//    the real 1000-col edge.
// --------------------------------------------------------------------------
__global__ __launch_bounds__(256) void logits_kernel(
    const float* __restrict__ ctx, const float* __restrict__ Woc,
    const float* __restrict__ boc, float* __restrict__ logits) {
  const int wave  = threadIdx.x >> 5;
  const int lane  = threadIdx.x & 31;
  const int nlane = lane & 15;
  const int rowbase = blockIdx.x * 64;

  const float* arow[4];
#pragma unroll
  for (int mt = 0; mt < 4; ++mt)
    arow[mt] = ctx + (size_t)(rowbase + mt * 16 + nlane) * DD;

  v8f acc[4][4];
  zero_acc<4, 4>(acc);
  const int nbase = blockIdx.y * 512 + wave * 64;
  wmma_kloop<4, 4, false>(arow, Woc, WOCLD, nbase, WOCLD, DD, acc);
  store_tile<4, 4>(logits, VOC, rowbase, BB * LL, nbase, VOC, boc, false,
                   false, acc);
}

// ---------------------------------------------------------------------------
extern "C" void kernel_launch(void* const* d_in, const int* in_sizes, int n_in,
                              void* d_out, int out_size, void* d_ws, size_t ws_size,
                              hipStream_t stream) {
  (void)in_sizes; (void)n_in; (void)out_size; (void)ws_size;

  const float* vf     = (const float*)d_in[0];
  const int*   ids    = (const int*)d_in[1];
  /* d_in[2] attention_mask: unused by the reference */
  const float* W_enc  = (const float*)d_in[3];
  const float* b_enc  = (const float*)d_in[4];
  const float* W_sel  = (const float*)d_in[5];
  const float* b_sel  = (const float*)d_in[6];
  const float* emb    = (const float*)d_in[7];
  const float* W_q    = (const float*)d_in[8];
  const float* W_k    = (const float*)d_in[9];
  const float* W_v    = (const float*)d_in[10];
  const float* b_q    = (const float*)d_in[11];
  const float* b_k    = (const float*)d_in[12];
  const float* b_v    = (const float*)d_in[13];
  const float* W_o    = (const float*)d_in[14];
  const float* b_o    = (const float*)d_in[15];
  const float* W_cls  = (const float*)d_in[16];
  const float* b_cls  = (const float*)d_in[17];

  float* out      = (float*)d_out;
  float* logits   = out;
  float* reps_out = out + (size_t)BB * LL * VOC;   // [32,6,512] after logits

  // workspace layout (fp32), ~71 MB total
  float* ws      = (float*)d_ws;
  float* scores  = ws;                        // 131072
  float* reps    = scores + (size_t)BB * TT;  // 98304
  float* kmat    = reps + (size_t)BB * KK * DD;
  float* vmat    = kmat + (size_t)BB * KK * DD;
  float* qbuf    = vmat + (size_t)BB * KK * DD;       // 8388608
  float* ctx     = qbuf + (size_t)BB * LL * DD;       // 8388608
  float* Woc     = ctx + (size_t)BB * LL * DD;        // 512*1024 padded
  float* boc     = Woc + (size_t)DD * WOCLD;          // 1024 (padded)
  int*   top_idx = (int*)(boc + 1024);                // 192

  // 1) encoder + frame scores (the 103-GFLOP pass)
  enc_score_kernel<<<dim3(BB * TT / 64), dim3(256), 0, stream>>>(
      vf, W_enc, b_enc, W_sel, b_sel, scores);
  // 2) top-6
  topk_kernel<<<dim3(BB), dim3(256), 0, stream>>>(scores, top_idx);
  // 3) reps (recompute 192 encoder rows)
  reps_kernel<<<dim3(BB * KK / 16, DD / 64), dim3(32), 0, stream>>>(
      vf, W_enc, b_enc, top_idx, reps, reps_out);
  // 4) k/v projections
  kv_kernel<<<dim3(BB * KK / 16, DD / 64, 2), dim3(32), 0, stream>>>(
      reps, W_k, b_k, W_v, b_v, kmat, vmat);
  // 5) q projection (gathered embedding rows)
  q_kernel<<<dim3(BB * LL / 64), dim3(256), 0, stream>>>(ids, emb, W_q, b_q, qbuf);
  // 6) fold W_o into classifier (padded-N store)
  woc_kernel<<<dim3(DD / 64, 2), dim3(256), 0, stream>>>(W_o, W_cls, Woc);
  boc_kernel<<<dim3(4), dim3(256), 0, stream>>>(b_o, W_cls, b_cls, boc);
  // 7) attention (K=6 softmax + context)
  attn_kernel<<<dim3(BB * LL, NH), dim3(32), 0, stream>>>(qbuf, kmat, vmat, ctx);
  // 8) logits
  logits_kernel<<<dim3(BB * LL / 64, 2), dim3(256), 0, stream>>>(ctx, Woc, boc, logits);
}
